// Lip2WavDecoder_28097676051301
// MI455X (gfx1250) — matrix operand order repaired and checked by reading, not compiled, not run on hardware
//
#include <hip/hip_runtime.h>
#include <stdint.h>

// ---------------------------------------------------------------------------
// Problem constants (match reference)
// ---------------------------------------------------------------------------
constexpr int B      = 32;
constexpr int TENC   = 90;
constexpr int E      = 512;
constexpr int NMELS  = 80;
constexpr int TDEC   = 300;
constexpr int AHID   = 1024;   // attention rnn hidden
constexpr int DHID   = 1024;   // decoder rnn hidden
constexpr int PN     = 256;    // prenet dim
constexpr int ADIM   = 128;    // attention dim
constexpr int NFILT  = 32;
constexpr int KCONV  = 31;
constexpr int PADC   = 15;

typedef __attribute__((ext_vector_type(8)))  __bf16 v8bf;
typedef __attribute__((ext_vector_type(16))) __bf16 v16bf;
typedef __attribute__((ext_vector_type(8)))  float  v8f;

#define DEVINL static __device__ __forceinline__

DEVINL uint16_t f2bf_bits(float f) {
  uint32_t u = __float_as_uint(f);
  uint32_t r = u + 0x7FFFu + ((u >> 16) & 1u);   // round-to-nearest-even
  return (uint16_t)(r >> 16);
}
DEVINL float sigf(float x) { return 1.0f / (1.0f + __expf(-x)); }

// ---------------------------------------------------------------------------
// WMMA fragment loaders (bf16, 16x16x32 shapes)
//   A (16x32, M on lanes): lane holds k = half*8 + e%8 + (e>=8 ? 16 : 0)
//   B (32x16, N on lanes): lane holds k = half*8 + e%8 + (e>=8 ? 16 : 0)
// ---------------------------------------------------------------------------
DEVINL v16bf ld_frag_a(const uint16_t* p) {  // p already includes half*8
  v8bf lo = *reinterpret_cast<const v8bf*>(p);
  v8bf hi = *reinterpret_cast<const v8bf*>(p + 16);
  v16bf r;
#pragma unroll
  for (int e = 0; e < 8; ++e) { r[e] = lo[e]; r[e + 8] = hi[e]; }
  return r;
}
DEVINL v16bf ld_frag_b(const uint16_t* p) {  // p already includes half*8
  v8bf lo = *reinterpret_cast<const v8bf*>(p);
  v8bf hi = *reinterpret_cast<const v8bf*>(p + 16);
  v16bf r;
#pragma unroll
  for (int e = 0; e < 8; ++e) { r[e] = lo[e]; r[e + 8] = hi[e]; }
  return r;
}

// One K-segment of an LSTM gate GEMM: activations A [32 x lda] (bf16 row-major),
// weights W [4096 x ldw] (bf16 row-major == B-matrix of A@W^T), starting at kofs.
struct Seg {
  const uint16_t* A;
  const uint16_t* W;
  int lda, ldw, kofs, ktiles;
};

DEVINL void gemm_seg(Seg sg, int lane16, int half, int wv, int n0, v8f* acc) {
  const uint16_t* a0 = sg.A + (size_t)lane16 * sg.lda + half * 8;
  const uint16_t* a1 = a0 + (size_t)16 * sg.lda;
  const uint16_t* w0 = sg.W + (size_t)(wv * 1024 + n0 + lane16) * sg.ldw + sg.kofs + half * 8;
  const uint16_t* w1 = w0 + (size_t)16 * sg.ldw;
  for (int kt = 0; kt < sg.ktiles; ++kt) {
    v16bf A0 = ld_frag_a(a0 + kt * 32);
    v16bf A1 = ld_frag_a(a1 + kt * 32);
    v16bf B0 = ld_frag_b(w0 + kt * 32);
    v16bf B1 = ld_frag_b(w1 + kt * 32);
    acc[0] = __builtin_amdgcn_wmma_f32_16x16x32_bf16(false, A0, false, B0, (short)0, acc[0], false, false);
    acc[1] = __builtin_amdgcn_wmma_f32_16x16x32_bf16(false, A0, false, B1, (short)0, acc[1], false, false);
    acc[2] = __builtin_amdgcn_wmma_f32_16x16x32_bf16(false, A1, false, B0, (short)0, acc[2], false, false);
    acc[3] = __builtin_amdgcn_wmma_f32_16x16x32_bf16(false, A1, false, B1, (short)0, acc[3], false, false);
  }
}

// ---------------------------------------------------------------------------
// Fused LSTM step: gates = sum_seg(A_s @ W_s^T) + b ; pointwise i,f,g,o update.
// Grid: 32 blocks (32 hidden cols each) x 128 threads (wave w computes gate w).
// ---------------------------------------------------------------------------
__global__ __launch_bounds__(128)
void lstm_step_kernel(Seg s0, Seg s1, Seg s2,
                      const float* __restrict__ bias,
                      float* __restrict__ c_state,
                      float* __restrict__ h_state,
                      uint16_t* __restrict__ h_bf) {
  __shared__ float s_g[4][32][32];
  const int tid = threadIdx.x;
  const int lane = tid & 31, wv = tid >> 5;
  const int lane16 = lane & 15, half = lane >> 4;
  const int n0 = blockIdx.x * 32;

  v8f acc[4];
#pragma unroll
  for (int i = 0; i < 4; ++i) acc[i] = v8f{0.f, 0.f, 0.f, 0.f, 0.f, 0.f, 0.f, 0.f};

  gemm_seg(s0, lane16, half, wv, n0, acc);
  gemm_seg(s1, lane16, half, wv, n0, acc);
  gemm_seg(s2, lane16, half, wv, n0, acc);

  // C/D layout: lane -> n = lane16, vgpr r -> m = i*16 + half*8 + r
#pragma unroll
  for (int i = 0; i < 2; ++i)
#pragma unroll
    for (int j = 0; j < 2; ++j)
#pragma unroll
      for (int r = 0; r < 8; ++r)
        s_g[wv][i * 16 + half * 8 + r][j * 16 + lane16] = acc[i * 2 + j][r];
  __syncthreads();

  for (int idx = tid; idx < 1024; idx += 128) {
    int bb = idx >> 5, nn = idx & 31;
    int col = n0 + nn;
    float gi = s_g[0][bb][nn] + bias[col];
    float gf = s_g[1][bb][nn] + bias[1024 + col];
    float gg = s_g[2][bb][nn] + bias[2048 + col];
    float go = s_g[3][bb][nn] + bias[3072 + col];
    float cp = c_state[bb * 1024 + col];
    float c2 = sigf(gf) * cp + sigf(gi) * tanhf(gg);
    float h2 = sigf(go) * tanhf(c2);
    c_state[bb * 1024 + col] = c2;
    h_state[bb * 1024 + col] = h2;
    h_bf[bb * 1024 + col] = f2bf_bits(h2);
  }
}

// ---------------------------------------------------------------------------
// Location-sensitive attention, one block per batch element.
// ---------------------------------------------------------------------------
__global__ __launch_bounds__(128)
void attention_kernel(const float* __restrict__ ah,
                      const float* __restrict__ memory,
                      const float* __restrict__ pm,
                      const float* __restrict__ conv_W,
                      const float* __restrict__ dense_W,
                      const float* __restrict__ v_W,
                      const float* __restrict__ query_W,
                      const int*   __restrict__ mem_len,
                      float* __restrict__ aw,
                      float* __restrict__ awc,
                      float* __restrict__ ctx,
                      uint16_t* __restrict__ ctx_bf,
                      float* __restrict__ align_out, int t) {
  __shared__ float s_aw[TENC], s_awc[TENC], s_q[ADIM];
  __shared__ float s_locf[NFILT][TENC];
  __shared__ float s_en[128], s_red[128];
  const int b = blockIdx.x, tid = threadIdx.x;

  if (tid < TENC) { s_aw[tid] = aw[b * TENC + tid]; s_awc[tid] = awc[b * TENC + tid]; }
  if (tid < ADIM) {
    float s = 0.f;
    const float* qr = query_W + (size_t)tid * AHID;
    const float* hr = ah + b * AHID;
    for (int k = 0; k < AHID; ++k) s += qr[k] * hr[k];
    s_q[tid] = s;
  }
  __syncthreads();

  for (int task = tid; task < NFILT * TENC; task += 128) {
    int nf = task / TENC, tt = task % TENC;
    float s = 0.f;
    const float* w0 = conv_W + nf * 2 * KCONV;
    for (int kk = 0; kk < KCONV; ++kk) {
      int idx = tt + kk - PADC;
      if (idx >= 0 && idx < TENC) s += w0[kk] * s_aw[idx] + w0[KCONV + kk] * s_awc[idx];
    }
    s_locf[nf][tt] = s;
  }
  __syncthreads();

  float en = -3.0e38f;
  if (tid < TENC) {
    float e = 0.f;
    const float* pmr = pm + ((size_t)(b * TENC) + tid) * ADIM;
    for (int ad = 0; ad < ADIM; ++ad) {
      float l2 = 0.f;
      const float* dw = dense_W + ad * NFILT;
#pragma unroll
      for (int nf = 0; nf < NFILT; ++nf) l2 += s_locf[nf][tid] * dw[nf];
      e += v_W[ad] * tanhf(s_q[ad] + l2 + pmr[ad]);
    }
    if (tid >= mem_len[b]) e = -1.0e9f;
    en = e;
  }
  s_en[tid] = en;
  s_red[tid] = en;
  __syncthreads();
  for (int s = 64; s > 0; s >>= 1) { if (tid < s) s_red[tid] = fmaxf(s_red[tid], s_red[tid + s]); __syncthreads(); }
  float mx = s_red[0];
  __syncthreads();
  float p = (tid < TENC) ? __expf(s_en[tid] - mx) : 0.f;
  s_red[tid] = p;
  __syncthreads();
  for (int s = 64; s > 0; s >>= 1) { if (tid < s) s_red[tid] += s_red[tid + s]; __syncthreads(); }
  float inv = 1.0f / s_red[0];
  __syncthreads();
  if (tid < TENC) {
    float a = p * inv;
    s_aw[tid] = a;
    aw[b * TENC + tid] = a;
    awc[b * TENC + tid] = s_awc[tid] + a;
    align_out[(size_t)b * TDEC * TENC + (size_t)t * TENC + tid] = a;
  }
  __syncthreads();
  for (int e = tid; e < E; e += 128) {
    float s = 0.f;
    const float* mr = memory + (size_t)b * TENC * E + e;
    for (int tt = 0; tt < TENC; ++tt) s += s_aw[tt] * mr[(size_t)tt * E];
    ctx[b * E + e] = s;
    ctx_bf[b * E + e] = f2bf_bits(s);
  }
}

// ---------------------------------------------------------------------------
// mel + gate projection; one block per batch element.
// ---------------------------------------------------------------------------
__global__ __launch_bounds__(128)
void proj_kernel(const float* __restrict__ dh, const float* __restrict__ ctx,
                 const float* __restrict__ proj_W, const float* __restrict__ proj_b,
                 const float* __restrict__ gate_W, const float* __restrict__ gate_b,
                 float* __restrict__ mel_out, float* __restrict__ gate_out, int t) {
  const int b = blockIdx.x, tid = threadIdx.x;
  if (tid <= NMELS) {
    const float* Wr = (tid < NMELS) ? (proj_W + (size_t)tid * (DHID + E)) : gate_W;
    float s = (tid < NMELS) ? proj_b[tid] : gate_b[0];
    const float* hr = dh + b * DHID;
    const float* cr = ctx + b * E;
    for (int k = 0; k < DHID; ++k) s += hr[k] * Wr[k];
    for (int k = 0; k < E; ++k) s += cr[k] * Wr[DHID + k];
    if (tid < NMELS) mel_out[(size_t)b * NMELS * TDEC + (size_t)tid * TDEC + t] = s;
    else             gate_out[b * TDEC + t] = s;
  }
}

// ---------------------------------------------------------------------------
// One-time setup kernels
// ---------------------------------------------------------------------------
__global__ void convert_bf16_kernel(const float* __restrict__ src, uint16_t* __restrict__ dst, long n) {
  long i = (long)blockIdx.x * blockDim.x + threadIdx.x;
  long stride = (long)gridDim.x * blockDim.x;
  for (; i < n; i += stride) dst[i] = f2bf_bits(src[i]);
}
__global__ void zero_f32_kernel(float* __restrict__ p, long n) {
  long i = (long)blockIdx.x * blockDim.x + threadIdx.x;
  long stride = (long)gridDim.x * blockDim.x;
  for (; i < n; i += stride) p[i] = 0.f;
}
// prenet layer 1: x1[t*32+b][j] = relu(dec_in[b][:80][t-1] @ W1[j]), zero at t=0
__global__ void prenet1_kernel(const float* __restrict__ dec_in, const float* __restrict__ W1,
                               float* __restrict__ x1) {
  long total = (long)TDEC * B * PN;
  long i = (long)blockIdx.x * blockDim.x + threadIdx.x;
  long stride = (long)gridDim.x * blockDim.x;
  for (; i < total; i += stride) {
    int j = (int)(i % PN);
    long rb = i / PN;
    int b = (int)(rb % B), t = (int)(rb / B);
    float s = 0.f;
    if (t > 0) {
      const float* dr = dec_in + (size_t)b * NMELS * TDEC + (t - 1);
      const float* wr = W1 + j * NMELS;
      for (int k = 0; k < NMELS; ++k) s += dr[(size_t)k * TDEC] * wr[k];
    }
    x1[i] = fmaxf(s, 0.f);
  }
}
// prenet layer 2 -> bf16 prenet outputs
__global__ void prenet2_kernel(const float* __restrict__ x1, const float* __restrict__ W2,
                               uint16_t* __restrict__ pbf) {
  long total = (long)TDEC * B * PN;
  long i = (long)blockIdx.x * blockDim.x + threadIdx.x;
  long stride = (long)gridDim.x * blockDim.x;
  for (; i < total; i += stride) {
    int j = (int)(i % PN);
    long rb = i / PN;
    const float* xr = x1 + rb * PN;
    const float* wr = W2 + j * PN;
    float s = 0.f;
    for (int k = 0; k < PN; ++k) s += xr[k] * wr[k];
    pbf[i] = f2bf_bits(fmaxf(s, 0.f));
  }
}
// processed_memory = memory @ memory_W^T  -> [B*TENC, ADIM]
__global__ void pm_kernel(const float* __restrict__ memory, const float* __restrict__ memW,
                          float* __restrict__ pm) {
  long total = (long)B * TENC * ADIM;
  long i = (long)blockIdx.x * blockDim.x + threadIdx.x;
  long stride = (long)gridDim.x * blockDim.x;
  for (; i < total; i += stride) {
    int ad = (int)(i % ADIM);
    long bt = i / ADIM;
    float s = 0.f;
    const float* mr = memory + bt * E;
    const float* wr = memW + (size_t)ad * E;
    for (int k = 0; k < E; ++k) s += mr[k] * wr[k];
    pm[i] = s;
  }
}

// ---------------------------------------------------------------------------
// Workspace layout (bytes, all 256B aligned)
// ---------------------------------------------------------------------------
constexpr size_t O_WAIH  = 0;                                  // attn Wih bf16 [4096x768]
constexpr size_t O_WAHH  = O_WAIH + (size_t)4096 * 768 * 2;    // attn Whh bf16 [4096x1024]
constexpr size_t O_WDIH  = O_WAHH + (size_t)4096 * 1024 * 2;   // dec Wih bf16 [4096x1536]
constexpr size_t O_WDHH  = O_WDIH + (size_t)4096 * 1536 * 2;   // dec Whh bf16 [4096x1024]
constexpr size_t O_X1    = O_WDHH + (size_t)4096 * 1024 * 2;   // prenet mid f32 [9600x256]
constexpr size_t O_PBF   = O_X1   + (size_t)TDEC * B * PN * 4; // prenet out bf16 [300x32x256]
constexpr size_t O_PM    = O_PBF  + (size_t)TDEC * B * PN * 2; // processed mem f32 [32x90x128]
constexpr size_t O_STATE = O_PM   + (size_t)B * TENC * ADIM * 4;
constexpr size_t O_AHS   = O_STATE;
constexpr size_t O_ACS   = O_AHS   + (size_t)B * AHID * 4;
constexpr size_t O_DHS   = O_ACS   + (size_t)B * AHID * 4;
constexpr size_t O_DCS   = O_DHS   + (size_t)B * DHID * 4;
constexpr size_t O_CTX   = O_DCS   + (size_t)B * DHID * 4;
constexpr size_t O_AW    = O_CTX   + (size_t)B * E * 4;
constexpr size_t O_AWC   = O_AW    + (size_t)B * TENC * 4;
constexpr size_t O_AHBF  = O_AWC   + (size_t)B * TENC * 4;
constexpr size_t O_DHBF  = O_AHBF  + (size_t)B * AHID * 2;
constexpr size_t O_CTXBF = O_DHBF  + (size_t)B * DHID * 2;
constexpr size_t O_END   = O_CTXBF + (size_t)B * E * 2;
constexpr long   STATE_F32 = (long)((O_END - O_STATE) / 4);

extern "C" void kernel_launch(void* const* d_in, const int* in_sizes, int n_in,
                              void* d_out, int out_size, void* d_ws, size_t ws_size,
                              hipStream_t stream) {
  (void)in_sizes; (void)n_in; (void)out_size; (void)ws_size;
  const float* memory  = (const float*)d_in[0];
  const float* dec_in  = (const float*)d_in[1];
  const float* pW1     = (const float*)d_in[2];
  const float* pW2     = (const float*)d_in[3];
  const float* aWih    = (const float*)d_in[4];
  const float* aWhh    = (const float*)d_in[5];
  const float* a_b     = (const float*)d_in[6];
  const float* qW      = (const float*)d_in[7];
  const float* mW      = (const float*)d_in[8];
  const float* convW   = (const float*)d_in[9];
  const float* denseW  = (const float*)d_in[10];
  const float* vW      = (const float*)d_in[11];
  const float* dWih    = (const float*)d_in[12];
  const float* dWhh    = (const float*)d_in[13];
  const float* d_b     = (const float*)d_in[14];
  const float* projW   = (const float*)d_in[15];
  const float* projb   = (const float*)d_in[16];
  const float* gateW   = (const float*)d_in[17];
  const float* gateb   = (const float*)d_in[18];
  const int*   memlen  = (const int*)d_in[19];

  char* ws = (char*)d_ws;
  uint16_t* Waih  = (uint16_t*)(ws + O_WAIH);
  uint16_t* Wahh  = (uint16_t*)(ws + O_WAHH);
  uint16_t* Wdih  = (uint16_t*)(ws + O_WDIH);
  uint16_t* Wdhh  = (uint16_t*)(ws + O_WDHH);
  float*    x1    = (float*)(ws + O_X1);
  uint16_t* pbf   = (uint16_t*)(ws + O_PBF);
  float*    pm    = (float*)(ws + O_PM);
  float*    ah    = (float*)(ws + O_AHS);
  float*    ac    = (float*)(ws + O_ACS);
  float*    dh    = (float*)(ws + O_DHS);
  float*    dc    = (float*)(ws + O_DCS);
  float*    ctx   = (float*)(ws + O_CTX);
  float*    aw    = (float*)(ws + O_AW);
  float*    awc   = (float*)(ws + O_AWC);
  uint16_t* ah_bf = (uint16_t*)(ws + O_AHBF);
  uint16_t* dh_bf = (uint16_t*)(ws + O_DHBF);
  uint16_t* ctx_bf= (uint16_t*)(ws + O_CTXBF);

  float* out       = (float*)d_out;
  float* mel_out   = out;                                  // [32][80][300]
  float* gate_out  = out + (size_t)B * NMELS * TDEC;       // [32][300]
  float* align_out = gate_out + (size_t)B * TDEC;          // [32][300][90]

  // ---- one-time setup ----
  convert_bf16_kernel<<<2048, 256, 0, stream>>>(aWih, Waih, (long)4096 * 768);
  convert_bf16_kernel<<<2048, 256, 0, stream>>>(aWhh, Wahh, (long)4096 * 1024);
  convert_bf16_kernel<<<2048, 256, 0, stream>>>(dWih, Wdih, (long)4096 * 1536);
  convert_bf16_kernel<<<2048, 256, 0, stream>>>(dWhh, Wdhh, (long)4096 * 1024);
  prenet1_kernel<<<4096, 256, 0, stream>>>(dec_in, pW1, x1);
  prenet2_kernel<<<4096, 256, 0, stream>>>(x1, pW2, pbf);
  pm_kernel<<<1440, 256, 0, stream>>>(memory, mW, pm);
  zero_f32_kernel<<<512, 256, 0, stream>>>((float*)(ws + O_STATE), STATE_F32);

  // ---- 300 sequential decoder steps ----
  for (int t = 0; t < TDEC; ++t) {
    // attention LSTM: [prenet_t | ctx] @ Wih^T + ah @ Whh^T
    Seg s0{ pbf + (size_t)t * B * PN, Waih, PN,   768,  0,   PN / 32 };
    Seg s1{ ctx_bf,                   Waih, E,    768,  PN,  E / 32 };
    Seg s2{ ah_bf,                    Wahh, AHID, 1024, 0,   AHID / 32 };
    lstm_step_kernel<<<32, 128, 0, stream>>>(s0, s1, s2, a_b, ac, ah, ah_bf);

    attention_kernel<<<32, 128, 0, stream>>>(ah, memory, pm, convW, denseW, vW, qW,
                                             memlen, aw, awc, ctx, ctx_bf, align_out, t);

    // decoder LSTM: [ah | ctx] @ Wih^T + dh @ Whh^T
    Seg d0{ ah_bf,  Wdih, AHID, 1536, 0,    AHID / 32 };
    Seg d1{ ctx_bf, Wdih, E,    1536, AHID, E / 32 };
    Seg d2{ dh_bf,  Wdhh, DHID, 1024, 0,    DHID / 32 };
    lstm_step_kernel<<<32, 128, 0, stream>>>(d0, d1, d2, d_b, dc, dh, dh_bf);

    proj_kernel<<<32, 128, 0, stream>>>(dh, ctx, projW, projb, gateW, gateb,
                                        mel_out, gate_out, t);
  }
}
